// SwinTransformerBlock_33182917328968
// MI455X (gfx1250) — matrix operand order repaired and checked by reading, not compiled, not run on hardware
//
#include <hip/hip_runtime.h>
#include <hip/hip_bf16.h>
#include <math.h>
#include <stdint.h>

#define USE_ASYNC_LDS 1

typedef _Float16 f16;
typedef _Float16 v16h __attribute__((ext_vector_type(16)));
typedef float    v8f  __attribute__((ext_vector_type(8)));

// ---------------- problem constants ----------------
constexpr int BATCH  = 2;
constexpr int RH = 32, RW = 32, RL = 16;
constexpr int DIM    = 384;
constexpr int HEADS  = 12;
constexpr int HD     = 32;            // head dim
constexpr int NT     = RH * RW * RL;  // 16384 tokens / batch
constexpr int NWIN   = 512;           // tokens per window (8*8*8)
constexpr int MROWS  = BATCH * NT;    // 32768 rows
constexpr int HIDDEN = 4 * DIM;       // 1536
constexpr int FEAT   = 2 * HD;        // 64 polarity features / head
constexpr int FDIM   = HEADS * FEAT;  // 768

// ---------------- small device math ----------------
__device__ inline float softplusf(float x) { return (x > 20.f) ? x : log1pf(expf(x)); }
__device__ inline float sigmoidf_(float x) { return 1.f / (1.f + expf(-x)); }
__device__ inline float relupowf(float x, float p) { return (x > 0.f) ? powf(x, p) : 0.f; }

// window-domain row -> token-domain row (window_reverse)
__device__ inline int win2tok(int r) {
  int win = r >> 9, n = r & 511;
  int b = win >> 5, wb = win & 31;
  int whi = wb >> 3, wwi = (wb >> 1) & 3, wli = wb & 1;
  int i = n >> 6, j = (n >> 3) & 7, k = n & 7;
  int h = whi * 8 + i, w = wwi * 8 + j, l = wli * 8 + k;
  return b * NT + (h * RW + w) * RL + l;
}
// token-domain row -> window-domain row (window_partition)
__device__ inline int tok2win(int t) {
  int b = t / NT, ti = t % NT;
  int h = ti >> 9, w = (ti >> 4) & 31, l = ti & 15;
  int win = b * 32 + (h >> 3) * 8 + (w >> 3) * 2 + (l >> 3);
  int n = (h & 7) * 64 + (w & 7) * 8 + (l & 7);
  return win * NWIN + n;
}

// ---------------- async copy helpers ----------------
__device__ inline void async_b128(const f16* gsrc, uint32_t lds_addr) {
#if USE_ASYNC_LDS
  asm volatile("global_load_async_to_lds_b128 %0, %1, off"
               :: "v"(lds_addr), "v"((unsigned long long)(uintptr_t)gsrc)
               : "memory");
#endif
}
__device__ inline void async_wait0() {
#if USE_ASYNC_LDS
  asm volatile("s_wait_asynccnt 0" ::: "memory");
#endif
}

// ---------------- f32 -> f16 weight convert ----------------
__global__ void f32_to_f16_k(const float* __restrict__ s, f16* __restrict__ d, int n) {
  int i = blockIdx.x * blockDim.x + threadIdx.x;
  if (i < n) d[i] = (f16)s[i];
}

// ---------------- LayerNorm (one block per token row) ----------------
__global__ void __launch_bounds__(128)
layernorm_k(const float* __restrict__ x, const float* __restrict__ gamma,
            const float* __restrict__ beta, f16* __restrict__ out, int remap) {
  const int t   = blockIdx.x;
  const int tid = threadIdx.x;
  const float* xr = x + (size_t)t * DIM;
  float v0 = xr[tid], v1 = xr[tid + 128], v2 = xr[tid + 256];
  __shared__ float rs[128], rq[128];
  rs[tid] = v0 + v1 + v2;
  rq[tid] = v0 * v0 + v1 * v1 + v2 * v2;
  __syncthreads();
  for (int off = 64; off; off >>= 1) {
    if (tid < off) { rs[tid] += rs[tid + off]; rq[tid] += rq[tid + off]; }
    __syncthreads();
  }
  __shared__ float smu, srstd;
  if (tid == 0) {
    float m = rs[0] * (1.f / DIM);
    float var = rq[0] * (1.f / DIM) - m * m;
    smu = m; srstd = rsqrtf(var + 1e-5f);
  }
  __syncthreads();
  const float mu = smu, rstd = srstd;
  const size_t orow = (size_t)(remap ? tok2win(t) : t);
  f16* o = out + orow * DIM;
  o[tid]       = (f16)((v0 - mu) * rstd * gamma[tid]       + beta[tid]);
  o[tid + 128] = (f16)((v1 - mu) * rstd * gamma[tid + 128] + beta[tid + 128]);
  o[tid + 256] = (f16)((v2 - mu) * rstd * gamma[tid + 256] + beta[tid + 256]);
}

// ---------------- GEMM epilogue modes ----------------
enum { EPI_QFEAT = 0, EPI_KVG = 1, EPI_PROJ = 2, EPI_FC1 = 3, EPI_FC2 = 4 };

struct EpiP {
  const float* bias;
  const float* pos_enc;   // [512*384]
  const float* scale_p;   // [384]
  const float* power_p;   // [384]
  float* qf;              // [MROWS*768]
  float* kcf;             // [MROWS*768]
  float* vout;            // [MROWS*384]
  float* gout;            // [MROWS*384]
  const float* xin;       // original x (token domain)
  float* x2;              // shortcut + attn (token domain)
  f16*   h1h;             // MLP hidden (f16)
  const float* x2r;       // x2 (read, token domain)
  float* dout;            // final output
};

template <int MODE>
__device__ inline void epilogue(float v, int r, int c, const EpiP& P) {
  if constexpr (MODE == EPI_QFEAT) {
    v += P.bias[c];
    v /= softplusf(P.scale_p[c]);
    const float p = 1.f + 4.f * sigmoidf_(P.power_p[c]);
    const int head = c >> 5, hd = c & 31;
    float* q = P.qf + (size_t)r * FDIM + head * FEAT + hd;
    q[0]  = relupowf(v, p);    // q_pos
    q[HD] = relupowf(-v, p);   // q_neg
  } else if constexpr (MODE == EPI_KVG) {
    if (c < DIM) {                               // K columns
      v += P.bias[c];
      v += P.pos_enc[(size_t)(r & (NWIN - 1)) * DIM + c];
      v /= softplusf(P.scale_p[c]);
      const float p = 1.f + 4.f * sigmoidf_(P.power_p[c]);
      const int head = c >> 5, hd = c & 31;
      float* k = P.kcf + (size_t)r * FDIM + head * FEAT + hd;
      k[0]  = relupowf(v, p);
      k[HD] = relupowf(-v, p);
    } else if (c < 2 * DIM) {                    // V columns
      P.vout[(size_t)r * DIM + (c - DIM)] = v + P.bias[c];
    } else {                                     // G columns
      P.gout[(size_t)r * DIM + (c - 2 * DIM)] = v + P.bias[c];
    }
  } else if constexpr (MODE == EPI_PROJ) {
    v += P.bias[c];
    const int t = win2tok(r);                    // window_reverse
    const size_t idx = (size_t)t * DIM + c;
    P.x2[idx] = P.xin[idx] + v;                  // residual 1
  } else if constexpr (MODE == EPI_FC1) {
    v += P.bias[c];
    const float gv = 0.5f * v * (1.f + erff(v * 0.70710678118654752f)); // exact GELU
    P.h1h[(size_t)r * HIDDEN + c] = (f16)gv;
  } else {                                       // EPI_FC2
    v += P.bias[c];
    const size_t idx = (size_t)r * DIM + c;
    P.dout[idx] = P.x2r[idx] + v;                // residual 2 -> output
  }
}

// ---------------- WMMA GEMM: D = A[MxK](f16) * W[NxK]^T(f16) + epilogue ----------------
// Block tile 128(M) x 128(N), 8 waves; each wave owns 16 rows x 128 cols = 8 WMMA accs.
// Double-buffered LDS; tile k+1 streamed with global_load_async_to_lds_b128 while
// tile k is consumed by v_wmma_f32_16x16x32_f16; s_wait_asynccnt 0 + barrier at tail.
template <int MODE>
__global__ void __launch_bounds__(256)
gemm_wmma_epi(const f16* __restrict__ A, const f16* __restrict__ W,
              int M, int N, int K, EpiP P) {
  __shared__ f16 sA[2][128 * 32];
  __shared__ f16 sB[2][128 * 32];
  const int tid  = threadIdx.x;
  const int lane = tid & 31;
  const int wid  = tid >> 5;
  const int rowBase = blockIdx.x * 128;
  const int colBase = blockIdx.y * 128;

  // ---- strength-reduced staging addresses (fixed per thread; only k advances) ----
  const int cr = tid >> 2;            // chunk row 0..63
  const int cs = (tid & 3) * 8;       // half offset within 32-half row
  const f16* gA0 = A + (size_t)(rowBase + cr) * K + cs;   // rows 0..63
  const f16* gA1 = gA0 + (size_t)64 * K;                  // rows 64..127
  const f16* gB0 = W + (size_t)(colBase + cr) * K + cs;   // cols 0..63
  const f16* gB1 = gB0 + (size_t)64 * K;                  // cols 64..127
  const int lofs0 = cr * 32 + cs;
  const int lofs1 = lofs0 + 64 * 32;

  auto stage = [&](int buf, int kh) {
#if USE_ASYNC_LDS
    async_b128(gA0 + kh, (uint32_t)(uintptr_t)&sA[buf][lofs0]);
    async_b128(gA1 + kh, (uint32_t)(uintptr_t)&sA[buf][lofs1]);
    async_b128(gB0 + kh, (uint32_t)(uintptr_t)&sB[buf][lofs0]);
    async_b128(gB1 + kh, (uint32_t)(uintptr_t)&sB[buf][lofs1]);
#else
    *(uint4*)&sA[buf][lofs0] = *(const uint4*)(gA0 + kh);
    *(uint4*)&sA[buf][lofs1] = *(const uint4*)(gA1 + kh);
    *(uint4*)&sB[buf][lofs0] = *(const uint4*)(gB0 + kh);
    *(uint4*)&sB[buf][lofs1] = *(const uint4*)(gB1 + kh);
#endif
  };

  v8f acc[8] = {};

  // wave32 WMMA fragment layouts as loaded here (contiguous k, valid since A and B match):
  //  A 16x32 f16: lanes 0-15 row M=lane, K=0..15; lanes 16-31 K=16..31
  //  B 32x16 f16: lanes 0-15 col N=lane, K=0..15; lanes 16-31 K=16..31
  const int kbA  = (lane < 16) ? 0 : 16;
  const int kbB  = (lane < 16) ? 0 : 16;
  const int arow = (wid * 16 + (lane & 15)) * 32;
  const int bcol = lane & 15;

  stage(0, 0);
  async_wait0();
  __syncthreads();

  int buf = 0;
  for (int k0 = 0; k0 < K; k0 += 32) {
    if (k0 + 32 < K) stage(buf ^ 1, k0 + 32);             // overlap copy with math
    if (k0 + 64 < K)                                      // global_prefetch_b8 hint
      __builtin_prefetch(gA0 + k0 + 64, 0, 1);

    union { uint4 u[2]; v16h h; } af;
    af.u[0] = *(const uint4*)&sA[buf][arow + kbA];
    af.u[1] = *(const uint4*)&sA[buf][arow + 8 + kbA];
#pragma unroll
    for (int j = 0; j < 8; ++j) {
      const v16h bf = *(const v16h*)&sB[buf][(j * 16 + bcol) * 32 + kbB];
      acc[j] = __builtin_amdgcn_wmma_f32_16x16x32_f16(
                   false, af.h, false, bf, (short)0, acc[j], false, false);
    }
    async_wait0();
    __syncthreads();
    buf ^= 1;
  }

  // D layout: VGPR r holds M=r (lanes 0-15) / M=r+8 (lanes 16-31), N=lane&15
  const int mrow = rowBase + wid * 16 + ((lane < 16) ? 0 : 8);
  const int ncol = colBase + (lane & 15);
#pragma unroll
  for (int j = 0; j < 8; ++j)
#pragma unroll
    for (int r = 0; r < 8; ++r)
      epilogue<MODE>(acc[j][r], mrow + r, ncol + j * 16, P);
}

// ---------------- linear attention core (one block per window*head) ----------------
__global__ void __launch_bounds__(512)
attn_core_k(const float* __restrict__ qf, const float* __restrict__ kcf,
            const float* __restrict__ v, float* __restrict__ xo) {
  const int win  = blockIdx.x / HEADS;
  const int head = blockIdx.x % HEADS;
  const int tid  = threadIdx.x;
  const size_t base = (size_t)win * NWIN;

  __shared__ float kv[FEAT][HD];   // [64][32]: d<16 -> branch1(v1), d>=16 -> branch2(v2)
  __shared__ float km[FEAT];

  // phase A: kv[f][d] = (1/N) * sum_n kc[n,f] * v[n, head*HD + d]
  float a0 = 0.f, a1 = 0.f, a2 = 0.f, a3 = 0.f;
  const int i0 = tid * 4;
  const int f0 = (i0 + 0) >> 5, d0 = (i0 + 0) & 31;
  const int f1 = (i0 + 1) >> 5, d1 = (i0 + 1) & 31;
  const int f2 = (i0 + 2) >> 5, d2 = (i0 + 2) & 31;
  const int f3 = (i0 + 3) >> 5, d3 = (i0 + 3) & 31;
  for (int n = 0; n < NWIN; ++n) {
    const float* kr = kcf + (base + n) * FDIM + head * FEAT;
    const float* vr = v   + (base + n) * DIM  + head * HD;
    a0 += kr[f0] * vr[d0];
    a1 += kr[f1] * vr[d1];
    a2 += kr[f2] * vr[d2];
    a3 += kr[f3] * vr[d3];
  }
  const float sc = 1.f / (float)NWIN;
  kv[f0][d0] = a0 * sc; kv[f1][d1] = a1 * sc;
  kv[f2][d2] = a2 * sc; kv[f3][d3] = a3 * sc;
  if (tid < FEAT) {
    float s = 0.f;
    for (int n = 0; n < NWIN; ++n) s += kcf[(base + n) * FDIM + head * FEAT + tid];
    km[tid] = s * sc;
  }
  __syncthreads();

  // phase B: per token
  const size_t row = base + tid;
  float qq[FEAT];
  const float* qr = qf + row * FDIM + head * FEAT;
#pragma unroll
  for (int f = 0; f < FEAT; ++f) qq[f] = qr[f];
  float zs = 1e-6f, zo = 1e-6f;
#pragma unroll
  for (int f = 0; f < FEAT; ++f) { zs += qq[f] * km[f]; zo += qq[f ^ HD] * km[f]; }
  zs = 1.f / zs; zo = 1.f / zo;
  float* out = xo + row * DIM + head * HD;
#pragma unroll
  for (int d = 0; d < HD / 2; ++d) {
    float xs = 0.f, xd = 0.f;
#pragma unroll
    for (int f = 0; f < FEAT; ++f) {
      xs += qq[f] * kv[f][d];                 // q_sim @ kv1
      xd += qq[f ^ HD] * kv[f][HD / 2 + d];   // q_opp @ kv2
    }
    out[d]          = xs * zs;
    out[HD / 2 + d] = xd * zo;
  }
}

// ---------------- 5x5x5 depthwise conv on V (one block per window*head) ----------------
__global__ void __launch_bounds__(512)
dwconv_k(const float* __restrict__ v, const float* __restrict__ w,
         const float* __restrict__ b, float* __restrict__ vv) {
  const int win  = blockIdx.x / HEADS;
  const int head = blockIdx.x % HEADS;
  const int n = threadIdx.x;
  const int i = n >> 6, j = (n >> 3) & 7, k = n & 7;
  const size_t base = (size_t)win * NWIN;
  for (int c = 0; c < HD; ++c) {
    float acc = b[c];
    for (int dh = 0; dh < 5; ++dh) {
      const int ih = i + dh - 2;
      if (ih < 0 || ih >= 8) continue;
      for (int dw = 0; dw < 5; ++dw) {
        const int jw = j + dw - 2;
        if (jw < 0 || jw >= 8) continue;
        for (int dl = 0; dl < 5; ++dl) {
          const int kl = k + dl - 2;
          if (kl < 0 || kl >= 8) continue;
          acc += v[(base + ih * 64 + jw * 8 + kl) * DIM + head * HD + c] *
                 w[((dh * 5 + dw) * 5 + dl) * HD + c];
        }
      }
    }
    vv[(base + n) * DIM + head * HD + c] = acc;
  }
}

// ---------------- combine: (attn + conv) * gate -> f16 ----------------
__global__ void combine_k(const float* __restrict__ xo, const float* __restrict__ vv,
                          const float* __restrict__ g, f16* __restrict__ awh, int total) {
  int i = blockIdx.x * blockDim.x + threadIdx.x;
  if (i < total) awh[i] = (f16)((xo[i] + vv[i]) * g[i]);
}

// ---------------- launch ----------------
extern "C" void kernel_launch(void* const* d_in, const int* in_sizes, int n_in,
                              void* d_out, int out_size, void* d_ws, size_t ws_size,
                              hipStream_t stream) {
  const float* x       = (const float*)d_in[0];
  const float* y       = (const float*)d_in[1];
  const float* w_qkvg  = (const float*)d_in[2];
  const float* b_qkvg  = (const float*)d_in[3];
  const float* w_proj  = (const float*)d_in[4];
  const float* b_proj  = (const float*)d_in[5];
  const float* dwc_w   = (const float*)d_in[6];
  const float* dwc_b   = (const float*)d_in[7];
  const float* power_p = (const float*)d_in[8];
  const float* scale_p = (const float*)d_in[9];
  const float* pos_enc = (const float*)d_in[10];
  const float* g1      = (const float*)d_in[11];
  const float* b1      = (const float*)d_in[12];
  const float* g2      = (const float*)d_in[13];
  const float* b2      = (const float*)d_in[14];
  const float* w_fc1   = (const float*)d_in[15];
  const float* b_fc1   = (const float*)d_in[16];
  const float* w_fc2   = (const float*)d_in[17];
  const float* b_fc2   = (const float*)d_in[18];
  float* out = (float*)d_out;

  // ---- workspace layout (with aliasing of dead buffers) ----
  char* ws = (char*)d_ws;
  const size_t SZ_H_MD = (size_t)MROWS * DIM * sizeof(f16);      // 25.2 MB
  const size_t SZ_F_MD = (size_t)MROWS * DIM * sizeof(float);    // 50.3 MB
  const size_t SZ_F_MF = (size_t)MROWS * FDIM * sizeof(float);   // 100.7 MB
  size_t o = 0;
  f16*   xnh  = (f16*)(ws + o);   o += SZ_H_MD;   // LN(x), window rows
  f16*   ynh  = (f16*)(ws + o);   o += SZ_H_MD;   // LN(y), window rows
  float* qfb  = (float*)(ws + o); o += SZ_F_MF;   // q features
  float* kcf  = (float*)(ws + o); o += SZ_F_MF;   // k features
  float* vbuf = (float*)(ws + o); o += SZ_F_MD;   // V
  float* gbuf = (float*)(ws + o); o += SZ_F_MD;   // gate
  float* xob  = (float*)(ws + o); o += SZ_F_MD;   // attention out
  float* vvb  = (float*)(ws + o); o += SZ_F_MD;   // conv out
  f16*   whq  = (f16*)(ws + o);   o += (size_t)4 * DIM * DIM * sizeof(f16);
  f16*   whp  = (f16*)(ws + o);   o += (size_t)DIM * DIM * sizeof(f16);
  f16*   whf1 = (f16*)(ws + o);   o += (size_t)HIDDEN * DIM * sizeof(f16);
  f16*   whf2 = (f16*)(ws + o);   o += (size_t)DIM * HIDDEN * sizeof(f16);
  // aliases of dead regions:
  f16*   awh = xnh;                              // gated attn (window rows, f16)
  float* x2  = qfb;                              // shortcut+attn (token rows)
  f16*   hh  = (f16*)((char*)qfb + SZ_F_MD);     // LN2 out (token rows, f16)
  f16*   h1h = (f16*)kcf;                        // MLP hidden (f16)

  // ---- 1) weights f32 -> f16 ----
  {
    int n;
    n = 4 * DIM * DIM;    f32_to_f16_k<<<(n + 255) / 256, 256, 0, stream>>>(w_qkvg, whq, n);
    n = DIM * DIM;        f32_to_f16_k<<<(n + 255) / 256, 256, 0, stream>>>(w_proj, whp, n);
    n = HIDDEN * DIM;     f32_to_f16_k<<<(n + 255) / 256, 256, 0, stream>>>(w_fc1, whf1, n);
    n = DIM * HIDDEN;     f32_to_f16_k<<<(n + 255) / 256, 256, 0, stream>>>(w_fc2, whf2, n);
  }

  // ---- 2) LN + window partition ----
  layernorm_k<<<MROWS, 128, 0, stream>>>(x, g1, b1, xnh, 1);
  layernorm_k<<<MROWS, 128, 0, stream>>>(y, g1, b1, ynh, 1);

  // ---- 3) q = LN(y) @ Wq^T -> polarity features ----
  {
    EpiP P = {};
    P.bias = b_qkvg; P.scale_p = scale_p; P.power_p = power_p; P.qf = qfb;
    gemm_wmma_epi<EPI_QFEAT><<<dim3(MROWS / 128, DIM / 128), 256, 0, stream>>>(
        ynh, whq, MROWS, DIM, DIM, P);
  }
  // ---- 4) [k|v|g] = LN(x) @ Wkvg^T (rows 384..1535 of w_qkvg) ----
  {
    EpiP P = {};
    P.bias = b_qkvg + DIM; P.pos_enc = pos_enc; P.scale_p = scale_p; P.power_p = power_p;
    P.kcf = kcf; P.vout = vbuf; P.gout = gbuf;
    gemm_wmma_epi<EPI_KVG><<<dim3(MROWS / 128, (3 * DIM) / 128), 256, 0, stream>>>(
        xnh, whq + (size_t)DIM * DIM, MROWS, 3 * DIM, DIM, P);
  }

  // ---- 5) linear attention + 6) depthwise conv + 7) gate ----
  attn_core_k<<<BATCH * 32 * HEADS, 512, 0, stream>>>(qfb, kcf, vbuf, xob);
  dwconv_k<<<BATCH * 32 * HEADS, 512, 0, stream>>>(vbuf, dwc_w, dwc_b, vvb);
  {
    int total = MROWS * DIM;
    combine_k<<<(total + 255) / 256, 256, 0, stream>>>(xob, vvb, gbuf, awh, total);
  }

  // ---- 8) proj GEMM + window_reverse + residual ----
  {
    EpiP P = {};
    P.bias = b_proj; P.xin = x; P.x2 = x2;
    gemm_wmma_epi<EPI_PROJ><<<dim3(MROWS / 128, DIM / 128), 256, 0, stream>>>(
        awh, whp, MROWS, DIM, DIM, P);
  }

  // ---- 9) LN2 ----
  layernorm_k<<<MROWS, 128, 0, stream>>>(x2, g2, b2, hh, 0);

  // ---- 10) fc1 + GELU ----
  {
    EpiP P = {};
    P.bias = b_fc1; P.h1h = h1h;
    gemm_wmma_epi<EPI_FC1><<<dim3(MROWS / 128, HIDDEN / 128), 256, 0, stream>>>(
        hh, whf1, MROWS, HIDDEN, DIM, P);
  }
  // ---- 11) fc2 + residual -> out ----
  {
    EpiP P = {};
    P.bias = b_fc2; P.x2r = x2; P.dout = out;
    gemm_wmma_epi<EPI_FC2><<<dim3(MROWS / 128, DIM / 128), 256, 0, stream>>>(
        h1h, whf2, MROWS, DIM, HIDDEN, P);
  }
}
